// DifferentiableDAG_4415226380959
// MI455X (gfx1250) — matrix operand order repaired and checked by reading, not compiled, not run on hardware
//
#include <hip/hip_runtime.h>

typedef __attribute__((ext_vector_type(16))) _Float16 v16h;
typedef __attribute__((ext_vector_type(8)))  _Float16 v8h;
typedef __attribute__((ext_vector_type(4)))  _Float16 v4h;
typedef __attribute__((ext_vector_type(8)))  float    v8f;
typedef __attribute__((ext_vector_type(4)))  float    v4f;

#define NVAR 128   // D
#define NHID 16    // H
#define NBAT 8192  // B

// ---------------------------------------------------------------------------
// Kernel 1: soft adjacency. Binary gumbel-softmax P(on) = sigmoid((2l+g1-g0)/T),
// zero diagonal. 16384 elements.
// ---------------------------------------------------------------------------
__global__ __launch_bounds__(256) void dag_adjacency_kernel(
    const float* __restrict__ logits, const float* __restrict__ temp,
    const float* __restrict__ gumbel, float* __restrict__ A)
{
    const int idx = blockIdx.x * 256 + threadIdx.x;   // v*128 + i
    const int v = idx >> 7;
    const int i = idx & (NVAR - 1);
    const float T  = temp[0];
    const float l  = logits[idx];
    const float g0 = gumbel[2 * idx + 0];
    const float g1 = gumbel[2 * idx + 1];
    const float z  = (2.0f * l + g1 - g0) / T;
    const float a  = 1.0f / (1.0f + expf(-z));
    A[idx] = (v == i) ? 0.0f : a;
}

// ---------------------------------------------------------------------------
// Kernel 1b: one-time exact split X = hi + lo into two f16 planes (row-major).
// 4 elements per thread.
// ---------------------------------------------------------------------------
__global__ __launch_bounds__(256) void dag_split_kernel(
    const float* __restrict__ X, _Float16* __restrict__ Xhi,
    _Float16* __restrict__ Xlo)
{
    const int t = blockIdx.x * 256 + threadIdx.x;     // 0 .. B*D/4-1
    const v4f x = *(const v4f*)(X + 4 * (size_t)t);
    v4h hi, lo;
    #pragma unroll
    for (int e = 0; e < 4; ++e) {
        const _Float16 h = (_Float16)x[e];
        hi[e] = h;
        lo[e] = (_Float16)(x[e] - (float)h);
    }
    *(v4h*)(Xhi + 4 * (size_t)t) = hi;
    *(v4h*)(Xlo + 4 * (size_t)t) = lo;
}

// ---------------------------------------------------------------------------
// Kernel 2 (fast path): per-variable masked MLP, split-f16 WMMA, A-operand
// loaded directly from pre-split f16 planes.
//   block <-> (variable i, 128-row slab). 8 waves, one 16-row tile each.
// ---------------------------------------------------------------------------
__global__ __launch_bounds__(256) void dag_mlp_f16_kernel(
    const _Float16* __restrict__ Xhi, const _Float16* __restrict__ Xlo,
    const float* __restrict__ A,
    const float* __restrict__ W1, const float* __restrict__ b1,
    const float* __restrict__ W2, const float* __restrict__ b2,
    float* __restrict__ recon)
{
    __shared__ _Float16 sWhi[NHID * NVAR];   // Weff hi, column-major [h][v]
    __shared__ _Float16 sWlo[NHID * NVAR];   // Weff lo
    __shared__ float    sHid[8 * 16 * 16];   // per-wave 16x16 hidden staging

    const int i   = blockIdx.x & (NVAR - 1);
    const int rb  = blockIdx.x >> 7;          // row-block (0..63), 128 rows each
    const int tid = threadIdx.x;

    // ---- build Weff_i = A[:,i] .* W1[i,:,:] in LDS as f16 hi/lo, col-major ----
    #pragma unroll
    for (int e = 0; e < 8; ++e) {
        const int idx = tid + 256 * e;            // 0..2047
        const int h = idx >> 7;
        const int v = idx & (NVAR - 1);
        const float w = A[v * NVAR + i] * W1[(i * NVAR + v) * NHID + h];
        const _Float16 hi = (_Float16)w;
        const _Float16 lo = (_Float16)(w - (float)hi);
        sWhi[h * NVAR + v] = hi;
        sWlo[h * NVAR + v] = lo;
    }
    __syncthreads();

    const int lane = tid & 31;
    const int wave = tid >> 5;
    const int lm   = lane & 15;   // A-operand row in tile; also C/D column (h)
    const int kh   = lane >> 4;   // lane-half selector in WMMA layouts

    const int row = rb * 128 + wave * 16 + lm;
    const _Float16* xrh = Xhi + (size_t)row * NVAR;
    const _Float16* xrl = Xlo + (size_t)row * NVAR;
    __builtin_prefetch(xrh, 0, 1);

    v8f c = {0.f, 0.f, 0.f, 0.f, 0.f, 0.f, 0.f, 0.f};

    // K = 128 in 4 chunks of 32; 3 WMMAs per chunk (hi*hi + hi*lo + lo*hi)
    #pragma unroll
    for (int chunk = 0; chunk < 4; ++chunk) {
        const int k0 = chunk * 32;

        // A-operand f16 layout (16x32): lane holds 16 consecutive K
        // (matching B): [k0+16kh, +8) -> a[0..7], [k0+16kh+8, +8) -> a[8..15]
        const v8h ah0 = *(const v8h*)(xrh + k0 + 16 * kh);
        const v8h ah1 = *(const v8h*)(xrh + k0 + 16 * kh + 8);
        const v8h al0 = *(const v8h*)(xrl + k0 + 16 * kh);
        const v8h al1 = *(const v8h*)(xrl + k0 + 16 * kh + 8);
        const v16h ahi = __builtin_shufflevector(ah0, ah1,
            0,1,2,3,4,5,6,7,8,9,10,11,12,13,14,15);
        const v16h alo = __builtin_shufflevector(al0, al1,
            0,1,2,3,4,5,6,7,8,9,10,11,12,13,14,15);

        // B-operand (32x16): lane holds 16 consecutive K for column n=lm,
        // K offset 16*kh -> contiguous in column-major LDS (32B aligned).
        const v16h bhi = *(const v16h*)(sWhi + lm * NVAR + k0 + 16 * kh);
        const v16h blo = *(const v16h*)(sWlo + lm * NVAR + k0 + 16 * kh);

        c = __builtin_amdgcn_wmma_f32_16x16x32_f16(false, ahi, false, bhi,
                                                   (short)0, c, false, false);
        c = __builtin_amdgcn_wmma_f32_16x16x32_f16(false, ahi, false, blo,
                                                   (short)0, c, false, false);
        c = __builtin_amdgcn_wmma_f32_16x16x32_f16(false, alo, false, bhi,
                                                   (short)0, c, false, false);
    }

    // bias + relu; stage hidden tile to LDS (C layout: row = vg + 8*kh, col = lm)
    const float bias = b1[i * NHID + lm];
    #pragma unroll
    for (int vg = 0; vg < 8; ++vg) {
        float hval = c[vg] + bias;
        hval = hval > 0.0f ? hval : 0.0f;
        sHid[wave * 256 + (vg + 8 * kh) * 16 + lm] = hval;
    }
    __syncthreads();

    // second layer: 16 lanes per wave, one output row each
    if (lane < 16) {
        float s = b2[i];
        #pragma unroll
        for (int h = 0; h < NHID; ++h)
            s += sHid[wave * 256 + lane * 16 + h] * W2[i * NHID + h];
        recon[(size_t)(rb * 128 + wave * 16 + lane) * NVAR + i] = s;
    }
}

// ---------------------------------------------------------------------------
// Kernel 2 (fallback, no workspace): identical but splits X in-register.
// ---------------------------------------------------------------------------
__global__ __launch_bounds__(256) void dag_mlp_kernel(
    const float* __restrict__ X,  const float* __restrict__ A,
    const float* __restrict__ W1, const float* __restrict__ b1,
    const float* __restrict__ W2, const float* __restrict__ b2,
    float* __restrict__ recon)
{
    __shared__ _Float16 sWhi[NHID * NVAR];
    __shared__ _Float16 sWlo[NHID * NVAR];
    __shared__ float    sHid[8 * 16 * 16];

    const int i   = blockIdx.x & (NVAR - 1);
    const int rb  = blockIdx.x >> 7;
    const int tid = threadIdx.x;

    #pragma unroll
    for (int e = 0; e < 8; ++e) {
        const int idx = tid + 256 * e;
        const int h = idx >> 7;
        const int v = idx & (NVAR - 1);
        const float w = A[v * NVAR + i] * W1[(i * NVAR + v) * NHID + h];
        const _Float16 hi = (_Float16)w;
        const _Float16 lo = (_Float16)(w - (float)hi);
        sWhi[h * NVAR + v] = hi;
        sWlo[h * NVAR + v] = lo;
    }
    __syncthreads();

    const int lane = tid & 31;
    const int wave = tid >> 5;
    const int lm   = lane & 15;
    const int kh   = lane >> 4;

    const int row = rb * 128 + wave * 16 + lm;
    const float* xrow = X + (size_t)row * NVAR;

    v8f c = {0.f, 0.f, 0.f, 0.f, 0.f, 0.f, 0.f, 0.f};

    #pragma unroll
    for (int chunk = 0; chunk < 4; ++chunk) {
        const int k0 = chunk * 32;
        const v4f xa = *(const v4f*)(xrow + k0 + 16 * kh);
        const v4f xb = *(const v4f*)(xrow + k0 + 16 * kh + 4);
        const v4f xc = *(const v4f*)(xrow + k0 + 16 * kh + 8);
        const v4f xd = *(const v4f*)(xrow + k0 + 16 * kh + 12);

        v16h ahi, alo;
        #pragma unroll
        for (int t = 0; t < 4; ++t) {
            const float x0 = xa[t]; const _Float16 h0 = (_Float16)x0;
            ahi[t]      = h0; alo[t]      = (_Float16)(x0 - (float)h0);
            const float x1 = xb[t]; const _Float16 h1 = (_Float16)x1;
            ahi[4 + t]  = h1; alo[4 + t]  = (_Float16)(x1 - (float)h1);
            const float x2 = xc[t]; const _Float16 h2 = (_Float16)x2;
            ahi[8 + t]  = h2; alo[8 + t]  = (_Float16)(x2 - (float)h2);
            const float x3 = xd[t]; const _Float16 h3 = (_Float16)x3;
            ahi[12 + t] = h3; alo[12 + t] = (_Float16)(x3 - (float)h3);
        }

        const v16h bhi = *(const v16h*)(sWhi + lm * NVAR + k0 + 16 * kh);
        const v16h blo = *(const v16h*)(sWlo + lm * NVAR + k0 + 16 * kh);

        c = __builtin_amdgcn_wmma_f32_16x16x32_f16(false, ahi, false, bhi,
                                                   (short)0, c, false, false);
        c = __builtin_amdgcn_wmma_f32_16x16x32_f16(false, ahi, false, blo,
                                                   (short)0, c, false, false);
        c = __builtin_amdgcn_wmma_f32_16x16x32_f16(false, alo, false, bhi,
                                                   (short)0, c, false, false);
    }

    const float bias = b1[i * NHID + lm];
    #pragma unroll
    for (int vg = 0; vg < 8; ++vg) {
        float hval = c[vg] + bias;
        hval = hval > 0.0f ? hval : 0.0f;
        sHid[wave * 256 + (vg + 8 * kh) * 16 + lm] = hval;
    }
    __syncthreads();

    if (lane < 16) {
        float s = b2[i];
        #pragma unroll
        for (int h = 0; h < NHID; ++h)
            s += sHid[wave * 256 + lane * 16 + h] * W2[i * NHID + h];
        recon[(size_t)(rb * 128 + wave * 16 + lane) * NVAR + i] = s;
    }
}

// ---------------------------------------------------------------------------
extern "C" void kernel_launch(void* const* d_in, const int* in_sizes, int n_in,
                              void* d_out, int out_size, void* d_ws, size_t ws_size,
                              hipStream_t stream) {
    (void)in_sizes; (void)n_in; (void)out_size;
    const float* X      = (const float*)d_in[0];
    const float* logits = (const float*)d_in[1];
    const float* temp   = (const float*)d_in[2];
    const float* gumbel = (const float*)d_in[3];
    const float* W1     = (const float*)d_in[4];
    const float* b1     = (const float*)d_in[5];
    const float* W2     = (const float*)d_in[6];
    const float* b2     = (const float*)d_in[7];

    float* out  = (float*)d_out;                  // recon [B, D] first
    float* Aout = out + (size_t)NBAT * NVAR;      // then A [D, D]

    dag_adjacency_kernel<<<(NVAR * NVAR) / 256, 256, 0, stream>>>(
        logits, temp, gumbel, Aout);

    const size_t planeBytes = (size_t)NBAT * NVAR * sizeof(_Float16); // 2 MB
    if (ws_size >= 2 * planeBytes) {
        _Float16* Xhi = (_Float16*)d_ws;
        _Float16* Xlo = Xhi + (size_t)NBAT * NVAR;
        dag_split_kernel<<<(NBAT * NVAR / 4) / 256, 256, 0, stream>>>(X, Xhi, Xlo);
        dag_mlp_f16_kernel<<<(NBAT / 128) * NVAR, 256, 0, stream>>>(
            Xhi, Xlo, Aout, W1, b1, W2, b2, out);
    } else {
        dag_mlp_kernel<<<(NBAT / 128) * NVAR, 256, 0, stream>>>(
            X, Aout, W1, b1, W2, b2, out);
    }
}
